// MetaCNNLSTM_79628693667894
// MI455X (gfx1250) — matrix operand order repaired and checked by reading, not compiled, not run on hardware
//
#include <hip/hip_runtime.h>
#include <hip/hip_bf16.h>
#include <cstdint>

#define B_    32
#define CIN_  64
#define T_    512
#define F_    256
#define K_    9
#define GN_G_ 8
#define H_    512
#define G4H_  2048
#define NCLS_ 50

typedef __attribute__((ext_vector_type(16))) _Float16 v16h;
typedef __attribute__((ext_vector_type(8)))  _Float16 h8;
typedef __attribute__((ext_vector_type(8)))  float    v8f;

union FragU { v16h v; h8 h[2]; };

// A-matrix 16x32 f16 fragment (per CDNA5 ISA 7.12.2):
// lane L: M = L&15 ; elements 0..7 -> K = kb + (L>>4)*8 + j ; 8..15 -> K = kb + 16 + (L>>4)*8 + (j-8)
__device__ __forceinline__ v16h load_fragA(const _Float16* __restrict__ base,
                                           size_t rowStride, int kbase) {
  const int lane = threadIdx.x & 31;
  const _Float16* p = base + (size_t)(lane & 15) * rowStride + (size_t)(kbase + ((lane >> 4) << 3));
  FragU u;
  u.h[0] = *(const h8*)(p);
  u.h[1] = *(const h8*)(p + 16);
  return u.v;
}

// B-matrix 32x16 f16 fragment: lane L: N = L&15 ; elements 0..7 -> K = kb + (L>>4)*8 + j ; 8..15 -> K = kb + 16 + (L>>4)*8 + (j-8)
// (a column of B == a row of the row-major N x K weight matrix, two runs of 8 contiguous halves, 16 apart)
__device__ __forceinline__ v16h load_fragB(const _Float16* __restrict__ base,
                                           size_t rowStride, int kbase) {
  const int lane = threadIdx.x & 31;
  const _Float16* p = base + (size_t)(lane & 15) * rowStride + (size_t)(kbase + ((lane >> 4) << 3));
  FragU u;
  u.h[0] = *(const h8*)(p);
  u.h[1] = *(const h8*)(p + 16);
  return u.v;
}

__device__ __forceinline__ v8f wmma_f16(v16h a, v16h b, v8f c) {
  return __builtin_amdgcn_wmma_f32_16x16x32_f16(false, a, false, b, (short)0, c, false, false);
}

// ---------------- Conv1d (direct; x (4MB) + w (0.6MB) live in 192MB L2) ----------------
__global__ __launch_bounds__(256) void conv1d_kernel(const float* __restrict__ x,
                                                     const float* __restrict__ w,
                                                     float* __restrict__ y) {
  int idx = blockIdx.x * 256 + threadIdx.x;      // B*F*T
  int t = idx & (T_ - 1);
  int f = (idx >> 9) & (F_ - 1);
  int b = idx >> 17;
  const float* xb = x + (size_t)b * CIN_ * T_;
  const float* wf = w + (size_t)f * CIN_ * K_;
  float s = 0.f;
  for (int c = 0; c < CIN_; ++c) {
    const float* xc = xb + c * T_;
    const float* wc = wf + c * K_;
#pragma unroll
    for (int k = 0; k < K_; ++k) {
      int ti = t + k - (K_ / 2);
      if (ti >= 0 && ti < T_) s += xc[ti] * wc[k];
    }
  }
  y[idx] = s;
}

// ---------------- GroupNorm: per (b,g) reduction ----------------
__global__ __launch_bounds__(256) void gn_stats_kernel(const float* __restrict__ y,
                                                       float* __restrict__ stats) {
  int bg = blockIdx.x;                 // b*8+g
  int b = bg >> 3, g = bg & 7;
  const float* base = y + ((size_t)b * F_ + (size_t)g * (F_ / GN_G_)) * T_;
  const int N = (F_ / GN_G_) * T_;     // 16384
  float s = 0.f, ss = 0.f;
  for (int i = threadIdx.x; i < N; i += 256) {
    float v = base[i];
    s += v; ss += v * v;
  }
  __shared__ float rs[256], rss[256];
  rs[threadIdx.x] = s; rss[threadIdx.x] = ss;
  __syncthreads();
  for (int off = 128; off > 0; off >>= 1) {
    if ((int)threadIdx.x < off) {
      rs[threadIdx.x]  += rs[threadIdx.x + off];
      rss[threadIdx.x] += rss[threadIdx.x + off];
    }
    __syncthreads();
  }
  if (threadIdx.x == 0) {
    float mean = rs[0] / (float)N;
    float var  = rss[0] / (float)N - mean * mean;
    stats[bg * 2]     = mean;
    stats[bg * 2 + 1] = rsqrtf(var + 1e-5f);
  }
}

// normalize + ReLU + transpose (B,F,T)->(B,T,F) + cvt f16
__global__ __launch_bounds__(256) void gn_apply_kernel(const float* __restrict__ y,
                                                       const float* __restrict__ stats,
                                                       const float* __restrict__ gamma,
                                                       const float* __restrict__ beta,
                                                       _Float16* __restrict__ x16) {
  int idx = blockIdx.x * 256 + threadIdx.x;      // B*F*T
  int t = idx & (T_ - 1);
  int f = (idx >> 9) & (F_ - 1);
  int b = idx >> 17;
  int bg = b * GN_G_ + (f >> 5);
  float v = (y[idx] - stats[bg * 2]) * stats[bg * 2 + 1] * gamma[f] + beta[f];
  v = fmaxf(v, 0.f);
  x16[((size_t)b * T_ + t) * F_ + f] = (_Float16)v;
}

// ---------------- generic f32 -> f16 ----------------
__global__ __launch_bounds__(256) void f32_to_f16_kernel(const float* __restrict__ in,
                                                         _Float16* __restrict__ out, int n) {
  int i = blockIdx.x * 256 + threadIdx.x;
  if (i < n) out[i] = (_Float16)in[i];
}

__global__ __launch_bounds__(256) void zero_f32_kernel(float* __restrict__ p, int n) {
  int i = blockIdx.x * 256 + threadIdx.x;
  if (i < n) p[i] = 0.f;
}

// ---------------- input-projection GEMM: G = X(M,K) * W(N,K)^T + bias ----------------
// block tile 64(M) x 256(N), 8 waves, wave = 32x64 (2 Mtiles x 4 Ntiles)
__global__ __launch_bounds__(256) void xproj_kernel(const _Float16* __restrict__ X, int Din,
                                                    const _Float16* __restrict__ W,
                                                    const float* __restrict__ bias,
                                                    float* __restrict__ G) {
  const int M0 = blockIdx.x * 64;
  const int N0 = blockIdx.y * 256;
  const int w    = threadIdx.x >> 5;
  const int lane = threadIdx.x & 31;
  const int row0 = M0 + (w >> 2) * 32;
  const int col0 = N0 + (w & 3) * 64;
  v8f acc[2][4] = {};
  for (int kb = 0; kb < Din; kb += 32) {
    if (kb + 32 < Din)
      __builtin_prefetch(X + (size_t)row0 * Din + kb + 32, 0, 0);  // global_prefetch_b8
    v16h a0 = load_fragA(X + (size_t)row0 * Din, (size_t)Din, kb);
    v16h a1 = load_fragA(X + (size_t)(row0 + 16) * Din, (size_t)Din, kb);
#pragma unroll
    for (int nt = 0; nt < 4; ++nt) {
      v16h bf = load_fragB(W + (size_t)(col0 + nt * 16) * Din, (size_t)Din, kb);
      acc[0][nt] = wmma_f16(a0, bf, acc[0][nt]);
      acc[1][nt] = wmma_f16(a1, bf, acc[1][nt]);
    }
  }
  const int mloc = 8 * (lane >> 4);
  const int n    = lane & 15;
#pragma unroll
  for (int mt = 0; mt < 2; ++mt) {
#pragma unroll
    for (int nt = 0; nt < 4; ++nt) {
      int col = col0 + nt * 16 + n;
      float bv = bias[col];
#pragma unroll
      for (int i = 0; i < 8; ++i) {
        int row = row0 + mt * 16 + mloc + i;
        G[(size_t)row * G4H_ + col] = acc[mt][nt][i] + bv;
      }
    }
  }
}

// ---------------- fused LSTM step: gates GEMM (WMMA) + cell update ----------------
// grid = H/64 blocks; each block: 64 hidden cols, all 4 gates, all 32 batch rows.
// 8 waves: wave w -> mtile (w>>2), gate (w&3), 4 Ntiles of 16. K = 512, step 32.
__global__ __launch_bounds__(256) void lstm_step_kernel(const float* __restrict__ G,
                                                        const _Float16* __restrict__ whh,
                                                        _Float16* __restrict__ seq,   // (B,T,H) f16
                                                        float* __restrict__ c_state,  // (B,H)
                                                        int t) {
  __shared__ float gbuf[4][B_][64];
  const int j0   = blockIdx.x * 64;
  const int w    = threadIdx.x >> 5;
  const int lane = threadIdx.x & 31;
  const int mt   = w >> 2;      // 0/1 -> batch rows 0-15 / 16-31
  const int gate = w & 3;       // i,f,g,o
  v8f acc[4] = {};

  if (t > 0) {  // h_{t-1}; at t==0 h==0 so recurrent term is zero
    const _Float16* abase = seq + (size_t)(t - 1) * H_ + (size_t)(mt * 16) * ((size_t)T_ * H_);
    for (int kb = 0; kb < H_; kb += 32) {
      v16h a = load_fragA(abase, (size_t)T_ * H_, kb);
#pragma unroll
      for (int nt = 0; nt < 4; ++nt) {
        const _Float16* bbase = whh + (size_t)(gate * H_ + j0 + nt * 16) * H_;
        v16h bf = load_fragB(bbase, (size_t)H_, kb);
        acc[nt] = wmma_f16(a, bf, acc[nt]);
      }
    }
  }

  // deposit gate pre-activations (+ precomputed x-projection incl. bias) into LDS
  {
    const int mloc = 8 * (lane >> 4);
    const int n    = lane & 15;
#pragma unroll
    for (int nt = 0; nt < 4; ++nt) {
#pragma unroll
      for (int i = 0; i < 8; ++i) {
        int m   = mt * 16 + mloc + i;                    // batch row
        int col = nt * 16 + n;                           // 0..63
        float gv = G[((size_t)m * T_ + t) * G4H_ + gate * H_ + j0 + col];
        gbuf[gate][m][col] = acc[nt][i] + gv;
      }
    }
  }
  __syncthreads();

  // pointwise cell update: 32*64 elems / 256 threads
  for (int e = threadIdx.x; e < B_ * 64; e += 256) {
    int m = e >> 6;
    int col = e & 63;
    int j = j0 + col;
    float ig = 1.f / (1.f + __expf(-gbuf[0][m][col]));
    float fg = 1.f / (1.f + __expf(-gbuf[1][m][col]));
    float gg = tanhf(gbuf[2][m][col]);
    float og = 1.f / (1.f + __expf(-gbuf[3][m][col]));
    float c  = fg * c_state[m * H_ + j] + ig * gg;
    c_state[m * H_ + j] = c;
    float h = og * tanhf(c);
    seq[(size_t)m * T_ * H_ + (size_t)t * H_ + j] = (_Float16)h;
  }
}

// ---------------- mean pool over time (from f16 h-sequence) ----------------
__global__ __launch_bounds__(256) void pool_kernel(const _Float16* __restrict__ seq,
                                                   float* __restrict__ feat) {
  int idx = blockIdx.x * 256 + threadIdx.x;     // B*H
  int h = idx & (H_ - 1);
  int b = idx >> 9;
  const _Float16* p = seq + (size_t)b * T_ * H_ + h;
  float s = 0.f;
  for (int t = 0; t < T_; ++t) s += (float)p[(size_t)t * H_];
  feat[idx] = s * (1.f / (float)T_);
}

// ---------------- classifier head ----------------
__global__ __launch_bounds__(256) void head_kernel(const float* __restrict__ feat,
                                                   const float* __restrict__ hw,
                                                   const float* __restrict__ hb,
                                                   float* __restrict__ out) {
  int idx = blockIdx.x * 256 + threadIdx.x;
  if (idx >= B_ * NCLS_) return;
  int c = idx % NCLS_;
  int b = idx / NCLS_;
  const float* fb = feat + (size_t)b * H_;
  const float* wc = hw + (size_t)c * H_;
  float s = hb[c];
  for (int k = 0; k < H_; ++k) s += fb[k] * wc[k];
  out[idx] = s;
}

extern "C" void kernel_launch(void* const* d_in, const int* in_sizes, int n_in,
                              void* d_out, int out_size, void* d_ws, size_t ws_size,
                              hipStream_t stream) {
  (void)in_sizes; (void)n_in; (void)out_size; (void)ws_size;
  const float* x_emg  = (const float*)d_in[0];
  const float* conv_w = (const float*)d_in[1];
  const float* gamma  = (const float*)d_in[2];
  const float* beta   = (const float*)d_in[3];
  const float* head_w = (const float*)d_in[4];
  const float* head_b = (const float*)d_in[5];
  const float* w_ih[3] = {(const float*)d_in[6], (const float*)d_in[9],  (const float*)d_in[12]};
  const float* w_hh[3] = {(const float*)d_in[7], (const float*)d_in[10], (const float*)d_in[13]};
  const float* bvec[3] = {(const float*)d_in[8], (const float*)d_in[11], (const float*)d_in[14]};
  float* out = (float*)d_out;

  // ---- workspace carve-out (all buffers written before read) ----
  char* ws = (char*)d_ws;
  size_t off = 0;
  auto take = [&](size_t bytes) -> void* {
    void* p = ws + off;
    off = (off + bytes + 255) & ~(size_t)255;
    return p;
  };
  float*    conv_out = (float*)   take(sizeof(float)    * (size_t)B_ * F_ * T_);     // 16 MiB
  float*    stats    = (float*)   take(sizeof(float)    * B_ * GN_G_ * 2);
  _Float16* x16_l1   = (_Float16*)take(sizeof(_Float16) * (size_t)B_ * T_ * F_);     // 8 MiB
  _Float16* wih16    = (_Float16*)take(sizeof(_Float16) * (size_t)G4H_ * H_);        // 2 MiB
  _Float16* whh16    = (_Float16*)take(sizeof(_Float16) * (size_t)G4H_ * H_);        // 2 MiB
  float*    Gbuf     = (float*)   take(sizeof(float)    * (size_t)B_ * T_ * G4H_);   // 128 MiB
  _Float16* seqA     = (_Float16*)take(sizeof(_Float16) * (size_t)B_ * T_ * H_);     // 16 MiB
  _Float16* seqB     = (_Float16*)take(sizeof(_Float16) * (size_t)B_ * T_ * H_);     // 16 MiB
  float*    c_state  = (float*)   take(sizeof(float)    * B_ * H_);
  float*    feat     = (float*)   take(sizeof(float)    * B_ * H_);

  // ---- frontend: conv + groupnorm + relu + transpose to (B,T,F) f16 ----
  conv1d_kernel<<<(B_ * F_ * T_) / 256, 256, 0, stream>>>(x_emg, conv_w, conv_out);
  gn_stats_kernel<<<B_ * GN_G_, 256, 0, stream>>>(conv_out, stats);
  gn_apply_kernel<<<(B_ * F_ * T_) / 256, 256, 0, stream>>>(conv_out, stats, gamma, beta, x16_l1);

  // ---- 3 stacked LSTM layers ----
  for (int l = 0; l < 3; ++l) {
    const int Din = (l == 0) ? F_ : H_;
    const _Float16* X = (l == 0) ? x16_l1 : ((l == 1) ? seqA : seqB);
    _Float16* seq_out = (l == 1) ? seqB : seqA;

    const int n_ih = G4H_ * Din;
    f32_to_f16_kernel<<<(n_ih + 255) / 256, 256, 0, stream>>>(w_ih[l], wih16, n_ih);
    f32_to_f16_kernel<<<(G4H_ * H_ + 255) / 256, 256, 0, stream>>>(w_hh[l], whh16, G4H_ * H_);

    dim3 gg((B_ * T_) / 64, G4H_ / 256);               // 256 x 8 blocks
    xproj_kernel<<<gg, 256, 0, stream>>>(X, Din, wih16, bvec[l], Gbuf);

    zero_f32_kernel<<<(B_ * H_) / 256, 256, 0, stream>>>(c_state, B_ * H_);
    for (int t = 0; t < T_; ++t)
      lstm_step_kernel<<<H_ / 64, 256, 0, stream>>>(Gbuf, whh16, seq_out, c_state, t);
  }

  // ---- pool + head ----
  pool_kernel<<<(B_ * H_) / 256, 256, 0, stream>>>(seqA, feat);  // layer-3 output lives in seqA
  head_kernel<<<(B_ * NCLS_ + 255) / 256, 256, 0, stream>>>(feat, head_w, head_b, out);
}
